// SwinTransformerBlock_11510512353812
// MI455X (gfx1250) — matrix operand order repaired and checked by reading, not compiled, not run on hardware
//
#include <hip/hip_runtime.h>
#include <hip/hip_bf16.h>
#include <math.h>
#include <stdint.h>

typedef __attribute__((ext_vector_type(16))) __bf16 v16bf;
typedef __attribute__((ext_vector_type(8)))  float  v8f;
typedef __attribute__((ext_vector_type(4)))  unsigned int u32x4;
typedef __attribute__((ext_vector_type(8)))  int i32x8;
typedef __attribute__((ext_vector_type(4)))  int i32x4;

#define TOKENS   131072
#define SEQ      16384
#define C_DIM    256
#define NHEADS   8
#define HEAD_DIM 32
#define WIN_SZ   64
#define SHIFT_SZ 32
#define HID      1024

// ---------------------------------------------------------------- WMMA helpers

__device__ __forceinline__ v8f wmma_bf16(v16bf a, v16bf b, v8f c) {
  return __builtin_amdgcn_wmma_f32_16x16x32_bf16(
      /*neg_a=*/false, a, /*neg_b=*/false, b,
      /*c_mod=*/(short)0, c, /*reuse_a=*/false, /*reuse_b=*/false);
}

// A fragment, 16x32 bf16, row-major source with leading dim ld.
__device__ __forceinline__ v16bf load_a16x32(const __bf16* base, int ld) {
  const int lane = threadIdx.x & 31;
  const int row  = lane & 15;
  const int kh   = (lane >> 4) << 3;
  union { v16bf v; uint4 u[2]; } f;
  f.u[0] = *(const uint4*)(base + (size_t)row * ld + kh);
  f.u[1] = *(const uint4*)(base + (size_t)row * ld + kh + 16);
  return f.v;
}

// B fragment (32x16) from M^T, M row-major with leading dim ld: B[k][n] = M[n][k].
__device__ __forceinline__ v16bf load_bT32x16(const __bf16* base, int ld) {
  const int lane = threadIdx.x & 31;
  const int row  = lane & 15;
  const int kb   = (lane >> 4) << 3;
  union { v16bf v; uint4 u[2]; } f;
  f.u[0] = *(const uint4*)(base + (size_t)row * ld + kb);
  f.u[1] = *(const uint4*)(base + (size_t)row * ld + kb + 16);
  return f.v;
}

// ------------------------------------------------- Tensor Data Mover (2-D tile)
// DMA a (rows x row_elems) bf16 tile (row stride stride_elems) from global to LDS.
// D# packed per CDNA5 ISA ch.8: group0 = {count/type/addr}, group1 = dims/strides.

__device__ __forceinline__ void tdm_load_2d(const void* gaddr, unsigned lds_off,
                                            unsigned row_elems, unsigned rows,
                                            unsigned stride_elems) {
  unsigned long long ga = (unsigned long long)(uintptr_t)gaddr;
  u32x4 g0;
  g0[0] = 1u;                                                 // count=1 (user D#)
  g0[1] = lds_off;                                            // lds_addr (bytes)
  g0[2] = (unsigned)(ga & 0xFFFFFFFFu);                       // global_addr[31:0]
  g0[3] = (unsigned)((ga >> 32) & 0x1FFFFFFu) | (2u << 30);   // addr[56:32] | type=2
  i32x8 g1;
  g1[0] = (int)(1u << 16);                                    // data_size=1 (2 bytes)
  g1[1] = (int)((row_elems & 0xFFFFu) << 16);                 // tensor_dim0[15:0]
  g1[2] = (int)(((row_elems >> 16) & 0xFFFFu) |
                ((rows & 0xFFFFu) << 16));                    // dim0[31:16] | dim1[15:0]
  g1[3] = (int)(((rows >> 16) & 0xFFFFu) |
                ((row_elems & 0xFFFFu) << 16));               // dim1[31:16] | tile_dim0
  g1[4] = (int)(rows & 0xFFFFu);                              // tile_dim1 (tile_dim2=0)
  g1[5] = (int)stride_elems;                                  // tensor_dim0_stride[31:0]
  g1[6] = 0;
  g1[7] = 0;
  i32x4 z4 = {0, 0, 0, 0};
  i32x8 z8 = {0, 0, 0, 0, 0, 0, 0, 0};
  __builtin_amdgcn_tensor_load_to_lds(g0, g1, z4, z4, z8, 0);
}

// ------------------------------------------------- weight cast + transpose
// in: K x N fp32 (row-major).  out: N x K bf16 (row-major) == W^T.

__global__ __launch_bounds__(256)
void cvt_transpose_kernel(const float* __restrict__ in, __bf16* __restrict__ outT,
                          int K, int N) {
  int idx = blockIdx.x * 256 + threadIdx.x;
  if (idx < K * N) {
    int k = idx / N, n = idx % N;
    outT[(size_t)n * K + k] = (__bf16)in[idx];
  }
}

// ---------------------------------------------------------------- LN1 + shift

__global__ __launch_bounds__(256)
void ln_shift_kernel(const float* __restrict__ x, const float* __restrict__ g,
                     const float* __restrict__ b, __bf16* __restrict__ xs) {
  const int wave = threadIdx.x >> 5;
  const int lane = threadIdx.x & 31;
  const size_t t = (size_t)blockIdx.x * 8 + wave;
  const int i = (int)(t % SEQ);
  __bf16* o = xs + t * C_DIM + lane * 8;
  if (i >= SEQ - SHIFT_SZ) {
    uint4 z = {0u, 0u, 0u, 0u};
    *(uint4*)o = z;
    return;
  }
  const float* xp = x + (t + SHIFT_SZ) * C_DIM + lane * 8;
  float4 f0 = ((const float4*)xp)[0];
  float4 f1 = ((const float4*)xp)[1];
  float v[8] = {f0.x, f0.y, f0.z, f0.w, f1.x, f1.y, f1.z, f1.w};
  float s = 0.f;
#pragma unroll
  for (int j = 0; j < 8; ++j) s += v[j];
#pragma unroll
  for (int d = 16; d >= 1; d >>= 1) s += __shfl_xor(s, d, 32);
  const float mu = s * (1.0f / C_DIM);
  float vs = 0.f;
#pragma unroll
  for (int j = 0; j < 8; ++j) { float dd = v[j] - mu; vs += dd * dd; }
#pragma unroll
  for (int d = 16; d >= 1; d >>= 1) vs += __shfl_xor(vs, d, 32);
  const float inv = rsqrtf(vs * (1.0f / C_DIM) + 1e-5f);
  float4 g0 = ((const float4*)(g + lane * 8))[0];
  float4 g1 = ((const float4*)(g + lane * 8))[1];
  float4 bb0 = ((const float4*)(b + lane * 8))[0];
  float4 bb1 = ((const float4*)(b + lane * 8))[1];
  float gg[8] = {g0.x, g0.y, g0.z, g0.w, g1.x, g1.y, g1.z, g1.w};
  float bv[8] = {bb0.x, bb0.y, bb0.z, bb0.w, bb1.x, bb1.y, bb1.z, bb1.w};
#pragma unroll
  for (int j = 0; j < 8; ++j)
    o[j] = (__bf16)((v[j] - mu) * inv * gg[j] + bv[j]);
}

// ------------------------------------------------- residual + unshift + LN2

__global__ __launch_bounds__(256)
void residual_ln2_kernel(const float* __restrict__ x, const float* __restrict__ attn,
                         const float* __restrict__ g, const float* __restrict__ b,
                         float* __restrict__ x1, __bf16* __restrict__ h) {
  const int wave = threadIdx.x >> 5;
  const int lane = threadIdx.x & 31;
  const size_t t = (size_t)blockIdx.x * 8 + wave;
  const int i = (int)(t % SEQ);
  const float* xp = x + t * C_DIM + lane * 8;
  float4 f0 = ((const float4*)xp)[0];
  float4 f1 = ((const float4*)xp)[1];
  float v[8] = {f0.x, f0.y, f0.z, f0.w, f1.x, f1.y, f1.z, f1.w};
  if (i >= SHIFT_SZ) {
    const float* ap = attn + (t - SHIFT_SZ) * C_DIM + lane * 8;
    float4 a0 = ((const float4*)ap)[0];
    float4 a1 = ((const float4*)ap)[1];
    v[0] += a0.x; v[1] += a0.y; v[2] += a0.z; v[3] += a0.w;
    v[4] += a1.x; v[5] += a1.y; v[6] += a1.z; v[7] += a1.w;
  }
  float* x1p = x1 + t * C_DIM + lane * 8;
  ((float4*)x1p)[0] = make_float4(v[0], v[1], v[2], v[3]);
  ((float4*)x1p)[1] = make_float4(v[4], v[5], v[6], v[7]);
  float s = 0.f;
#pragma unroll
  for (int j = 0; j < 8; ++j) s += v[j];
#pragma unroll
  for (int d = 16; d >= 1; d >>= 1) s += __shfl_xor(s, d, 32);
  const float mu = s * (1.0f / C_DIM);
  float vs = 0.f;
#pragma unroll
  for (int j = 0; j < 8; ++j) { float dd = v[j] - mu; vs += dd * dd; }
#pragma unroll
  for (int d = 16; d >= 1; d >>= 1) vs += __shfl_xor(vs, d, 32);
  const float inv = rsqrtf(vs * (1.0f / C_DIM) + 1e-5f);
  float4 g0 = ((const float4*)(g + lane * 8))[0];
  float4 g1 = ((const float4*)(g + lane * 8))[1];
  float4 bb0 = ((const float4*)(b + lane * 8))[0];
  float4 bb1 = ((const float4*)(b + lane * 8))[1];
  float gg[8] = {g0.x, g0.y, g0.z, g0.w, g1.x, g1.y, g1.z, g1.w};
  float bv[8] = {bb0.x, bb0.y, bb0.z, bb0.w, bb1.x, bb1.y, bb1.z, bb1.w};
  __bf16* o = h + t * C_DIM + lane * 8;
#pragma unroll
  for (int j = 0; j < 8; ++j)
    o[j] = (__bf16)((v[j] - mu) * inv * gg[j] + bv[j]);
}

// ---------------------------------------------------------------- GEMM (WMMA)
// out[M,N] = A[M,K](bf16) @ W[K,N] + bias, W supplied TRANSPOSED (Wt: N x K bf16).
// Block = 128 threads = 4 waves; block tile 64(M) x 128(N); wave tile 32 x 64.
// STAGE_B: TDM-stage the 128 x K weight tile in LDS (requires K == 256).
// Epilogue: 0 store bf16 | 1 GELU->bf16 | 2 store f32 | 3 +res -> f32

template <int EPI, bool STAGE_B>
__global__ __launch_bounds__(128)
void gemm_bf16_kernel(const __bf16* __restrict__ A, const __bf16* __restrict__ Wt,
                      const float* __restrict__ bias, const float* __restrict__ res,
                      void* __restrict__ outp, int M, int N, int K) {
  __shared__ __bf16 Bl[STAGE_B ? 128 * 256 : 16];
  const int wave  = threadIdx.x >> 5;
  const int lane  = threadIdx.x & 31;
  const int waveM = blockIdx.y * 64 + (wave >> 1) * 32;
  const int waveN = blockIdx.x * 128 + (wave & 1) * 64;

  if (STAGE_B) {
    if (wave == 0) {
      tdm_load_2d(Wt + (size_t)blockIdx.x * 128 * K,
                  (unsigned)(uintptr_t)&Bl[0], (unsigned)K, 128u, (unsigned)K);
      __builtin_amdgcn_s_wait_tensorcnt(0);
    }
    __syncthreads();
  }

  v8f acc[2][4] = {};
  for (int kk = 0; kk < K; kk += 32) {
    v16bf a0 = load_a16x32(A + (size_t)waveM * K + kk, K);
    v16bf a1 = load_a16x32(A + (size_t)(waveM + 16) * K + kk, K);
    v16bf bfr[4];
#pragma unroll
    for (int nt = 0; nt < 4; ++nt) {
      const __bf16* bb = STAGE_B
          ? &Bl[0] + (size_t)((wave & 1) * 64 + nt * 16) * K + kk
          : Wt + (size_t)(waveN + nt * 16) * K + kk;
      bfr[nt] = load_bT32x16(bb, K);
    }
#pragma unroll
    for (int nt = 0; nt < 4; ++nt) {
      acc[0][nt] = wmma_bf16(a0, bfr[nt], acc[0][nt]);
      acc[1][nt] = wmma_bf16(a1, bfr[nt], acc[1][nt]);
    }
  }
  const int rrow = (lane >> 4) << 3;
  const int c0   = lane & 15;
#pragma unroll
  for (int mt = 0; mt < 2; ++mt)
#pragma unroll
    for (int nt = 0; nt < 4; ++nt)
#pragma unroll
      for (int r = 0; r < 8; ++r) {
        const int row = waveM + mt * 16 + rrow + r;
        const int col = waveN + nt * 16 + c0;
        float v = acc[mt][nt][r] + bias[col];
        if (EPI == 1) v = 0.5f * v * (1.0f + erff(v * 0.70710678118654752f));
        if (EPI == 3) v += res[(size_t)row * N + col];
        if (EPI <= 1) ((__bf16*)outp)[(size_t)row * N + col] = (__bf16)v;
        else          ((float*)outp)[(size_t)row * N + col] = v;
      }
}

// ------------------------------------------------------------- window attention
// One block per (window, head). 4 waves; wave w owns rows 16w..16w+15.
// K tile DMA'd to LDS by the TDM; V^T staged in LDS; bias column staged in LDS.

__global__ __launch_bounds__(128)
void window_attn_kernel(const __bf16* __restrict__ qkv, const float* __restrict__ rpb,
                        __bf16* __restrict__ out) {
  __shared__ __bf16 Pl[WIN_SZ * WIN_SZ];     // 8 KB
  __shared__ __bf16 Vt[HEAD_DIM * WIN_SZ];   // 4 KB : Vt[c][k] = V[k][c]
  __shared__ __bf16 Kl[WIN_SZ * HEAD_DIM];   // 4 KB : K tile (row-major 64x32)
  __shared__ float  biasc[128];              // rpb column for this head
  const int win  = blockIdx.x;
  const int h    = blockIdx.y;
  const int wave = threadIdx.x >> 5;
  const int lane = threadIdx.x & 31;
  const int ld   = 3 * C_DIM;
  const __bf16* qb = qkv + (size_t)win * WIN_SZ * ld + h * HEAD_DIM;
  const __bf16* kb = qb + C_DIM;
  const __bf16* vb = qb + 2 * C_DIM;

  // TDM: DMA the 64x32 (stride 768) K tile into LDS
  if (wave == 0)
    tdm_load_2d(kb, (unsigned)(uintptr_t)&Kl[0], HEAD_DIM, WIN_SZ, (unsigned)ld);

  // bias column for this head
  if (threadIdx.x < 2 * WIN_SZ - 1)
    biasc[threadIdx.x] = rpb[threadIdx.x * NHEADS + h];

  // cooperative coalesced load of V, transposed store into LDS
  {
    const int tid   = threadIdx.x;     // 0..127
    const int vrow  = tid >> 1;        // k index 0..63
    const int halfo = (tid & 1) * 16;  // col half
    union { uint4 u[2]; __bf16 e[16]; } buf;
    const __bf16* vp = vb + (size_t)vrow * ld + halfo;
    buf.u[0] = ((const uint4*)vp)[0];
    buf.u[1] = ((const uint4*)vp)[1];
#pragma unroll
    for (int j = 0; j < 16; ++j) Vt[(halfo + j) * WIN_SZ + vrow] = buf.e[j];
  }

  if (wave == 0) __builtin_amdgcn_s_wait_tensorcnt(0);
  __syncthreads();

  // S = q @ k^T  (K = head_dim = 32, one WMMA per 16x16 tile); k^T from LDS
  v16bf aq = load_a16x32(qb + (size_t)(wave * 16) * ld, ld);
  v8f s[4] = {};
#pragma unroll
  for (int nt = 0; nt < 4; ++nt) {
    v16bf bk = load_bT32x16(&Kl[0] + (size_t)(nt * 16) * HEAD_DIM, HEAD_DIM);
    s[nt] = wmma_bf16(aq, bk, s[nt]);
  }

  const int rrow = (lane >> 4) << 3;
  const int c0   = lane & 15;
  const float scale = 0.17677669529663687f;  // 32^-0.5
  float p[4][8];
#pragma unroll
  for (int nt = 0; nt < 4; ++nt)
#pragma unroll
    for (int r = 0; r < 8; ++r) {
      const int i = wave * 16 + rrow + r;
      const int j = nt * 16 + c0;
      p[nt][r] = s[nt][r] * scale + biasc[i - j + WIN_SZ - 1];
    }

  // row softmax: row lives across one 16-lane half (masks 1,2,4,8 stay in-half)
#pragma unroll
  for (int r = 0; r < 8; ++r) {
    float m = fmaxf(fmaxf(p[0][r], p[1][r]), fmaxf(p[2][r], p[3][r]));
#pragma unroll
    for (int d = 8; d >= 1; d >>= 1) m = fmaxf(m, __shfl_xor(m, d, 32));
    float sum = 0.f;
#pragma unroll
    for (int nt = 0; nt < 4; ++nt) { p[nt][r] = __expf(p[nt][r] - m); sum += p[nt][r]; }
#pragma unroll
    for (int d = 8; d >= 1; d >>= 1) sum += __shfl_xor(sum, d, 32);
    const float inv = 1.0f / sum;
#pragma unroll
    for (int nt = 0; nt < 4; ++nt) p[nt][r] *= inv;
  }

  // stage P (bf16) in LDS, then re-fragment as WMMA A operands
#pragma unroll
  for (int nt = 0; nt < 4; ++nt)
#pragma unroll
    for (int r = 0; r < 8; ++r)
      Pl[(wave * 16 + rrow + r) * WIN_SZ + nt * 16 + c0] = (__bf16)p[nt][r];
  __syncthreads();

  v8f o[2] = {};
#pragma unroll
  for (int kc = 0; kc < 2; ++kc) {
    const int row = lane & 15;
    const int kh  = (lane >> 4) << 3;
    union { v16bf v; uint4 u[2]; } fa;
    fa.u[0] = *(const uint4*)&Pl[(wave * 16 + row) * WIN_SZ + kc * 32 + kh];
    fa.u[1] = *(const uint4*)&Pl[(wave * 16 + row) * WIN_SZ + kc * 32 + kh + 16];
#pragma unroll
    for (int nt = 0; nt < 2; ++nt) {
      const int kbb = (lane >> 4) << 3;
      union { v16bf v; uint4 u[2]; } fb;
      fb.u[0] = *(const uint4*)&Vt[(nt * 16 + c0) * WIN_SZ + kc * 32 + kbb];
      fb.u[1] = *(const uint4*)&Vt[(nt * 16 + c0) * WIN_SZ + kc * 32 + kbb + 16];
      o[nt] = wmma_bf16(fa.v, fb.v, o[nt]);
    }
  }
#pragma unroll
  for (int nt = 0; nt < 2; ++nt)
#pragma unroll
    for (int r = 0; r < 8; ++r) {
      const int row = win * WIN_SZ + wave * 16 + rrow + r;
      const int col = h * HEAD_DIM + nt * 16 + c0;
      out[(size_t)row * C_DIM + col] = (__bf16)o[nt][r];
    }
}

// ---------------------------------------------------------------- launch

extern "C" void kernel_launch(void* const* d_in, const int* in_sizes, int n_in,
                              void* d_out, int out_size, void* d_ws, size_t ws_size,
                              hipStream_t stream) {
  (void)in_sizes; (void)n_in; (void)out_size; (void)ws_size;
  const float* x      = (const float*)d_in[0];
  const float* g1     = (const float*)d_in[1];
  const float* b1     = (const float*)d_in[2];
  const float* qkv_w  = (const float*)d_in[3];
  const float* qkv_b  = (const float*)d_in[4];
  const float* rpb    = (const float*)d_in[5];
  const float* proj_w = (const float*)d_in[6];
  const float* proj_b = (const float*)d_in[7];
  const float* g2     = (const float*)d_in[8];
  const float* b2     = (const float*)d_in[9];
  const float* fc1_w  = (const float*)d_in[10];
  const float* fc1_b  = (const float*)d_in[11];
  const float* fc2_w  = (const float*)d_in[12];
  const float* fc2_b  = (const float*)d_in[13];
  float* out = (float*)d_out;

  char* ws = (char*)d_ws;
  auto alloc = [&](size_t bytes) {
    char* p = ws;
    ws += (bytes + 255) & ~(size_t)255;
    return p;
  };
  __bf16* qkv_wt  = (__bf16*)alloc((size_t)C_DIM * 3 * C_DIM * 2);   // (3C x C)
  __bf16* proj_wt = (__bf16*)alloc((size_t)C_DIM * C_DIM * 2);       // (C x C)
  __bf16* fc1_wt  = (__bf16*)alloc((size_t)C_DIM * HID * 2);         // (HID x C)
  __bf16* fc2_wt  = (__bf16*)alloc((size_t)HID * C_DIM * 2);         // (C x HID)
  __bf16* xs      = (__bf16*)alloc((size_t)TOKENS * C_DIM * 2);
  __bf16* qkvb    = (__bf16*)alloc((size_t)TOKENS * 3 * C_DIM * 2);
  __bf16* attnb   = (__bf16*)alloc((size_t)TOKENS * C_DIM * 2);
  float*  projf   = (float*) alloc((size_t)TOKENS * C_DIM * 4);
  float*  x1      = (float*) alloc((size_t)TOKENS * C_DIM * 4);
  __bf16* hb      = (__bf16*)alloc((size_t)TOKENS * C_DIM * 2);
  __bf16* a1      = (__bf16*)alloc((size_t)TOKENS * HID * 2);

  // weight cast + transpose (W^T used as WMMA B operand source)
  cvt_transpose_kernel<<<(C_DIM * 3 * C_DIM + 255) / 256, 256, 0, stream>>>(qkv_w, qkv_wt, C_DIM, 3 * C_DIM);
  cvt_transpose_kernel<<<(C_DIM * C_DIM + 255) / 256, 256, 0, stream>>>(proj_w, proj_wt, C_DIM, C_DIM);
  cvt_transpose_kernel<<<(C_DIM * HID + 255) / 256, 256, 0, stream>>>(fc1_w, fc1_wt, C_DIM, HID);
  cvt_transpose_kernel<<<(HID * C_DIM + 255) / 256, 256, 0, stream>>>(fc2_w, fc2_wt, HID, C_DIM);

  // LN1 + shift
  ln_shift_kernel<<<TOKENS / 8, 256, 0, stream>>>(x, g1, b1, xs);

  // QKV projection (TDM-staged weights, K=256)
  gemm_bf16_kernel<0, true><<<dim3(3 * C_DIM / 128, TOKENS / 64), 128, 0, stream>>>(
      xs, qkv_wt, qkv_b, nullptr, qkvb, TOKENS, 3 * C_DIM, C_DIM);

  // window attention
  window_attn_kernel<<<dim3(TOKENS / WIN_SZ, NHEADS), 128, 0, stream>>>(qkvb, rpb, attnb);

  // output projection (TDM-staged weights, K=256)
  gemm_bf16_kernel<2, true><<<dim3(C_DIM / 128, TOKENS / 64), 128, 0, stream>>>(
      attnb, proj_wt, proj_b, nullptr, projf, TOKENS, C_DIM, C_DIM);

  // residual + unshift + LN2
  residual_ln2_kernel<<<TOKENS / 8, 256, 0, stream>>>(x, projf, g2, b2, x1, hb);

  // MLP
  gemm_bf16_kernel<1, true><<<dim3(HID / 128, TOKENS / 64), 128, 0, stream>>>(
      hb, fc1_wt, fc1_b, nullptr, a1, TOKENS, HID, C_DIM);
  gemm_bf16_kernel<3, false><<<dim3(C_DIM / 128, TOKENS / 64), 128, 0, stream>>>(
      a1, fc2_wt, fc2_b, x1, out, TOKENS, C_DIM, HID);
}
